// RPN_89876485636848
// MI455X (gfx1250) — matrix operand order repaired and checked by reading, not compiled, not run on hardware
//
#include <hip/hip_runtime.h>

typedef __attribute__((ext_vector_type(16))) __bf16 v16bf;
typedef __attribute__((ext_vector_type(16))) unsigned short v16u;
typedef __attribute__((ext_vector_type(8))) float v8f;

#define TOPK 1000
#define NANCH 59850
// out layout (floats): obj[59850] | proposals[59850*4] | top_scores[1000] | top_boxes[4000] | anchors[59850*4]
#define OUT_PROP 59850
#define OUT_TS   299250
#define OUT_TB   300250
#define OUT_ANC  304250

// workspace layout (bytes)
static constexpr size_t WS_WPACK  = 0;        // 72*16*32*16 u16 = 1,179,648
static constexpr size_t WS_HPACK  = 1179648;  // 8*32*16 u16 = 8,192
static constexpr size_t WS_SCORES = 1187840;  // 59850 f32
static constexpr size_t WS_HIST   = 1427456;  // 4096 u32
static constexpr size_t WS_META   = 1443840;  // 8 u32 (cutbin, need, slotCtr, tieCtr)
static constexpr size_t WS_PSC    = 1443904;  // 1024 f32
static constexpr size_t WS_PIDX   = 1448000;  // 1024 i32

__device__ __forceinline__ unsigned short f2bf(float f) {
  unsigned int u = __float_as_uint(f);
  u += 0x7FFFu + ((u >> 16) & 1u);   // round-to-nearest-even
  return (unsigned short)(u >> 16);
}

// ---------------------------------------------------------------------------
// K0: pack conv weights (OIHW f32) into bf16 WMMA B-fragments, K = khw*256+ic
//     B layout (16-bit, 32x16): lane L -> N=L%16, K=(L>=16?8:0)+(e<8?e:e+8) (e=0..15)
//     wpack index = ((s*16 + nt)*32 + lane)*16 + e  -->  for a fixed wave the
//     per-s stride is a constant 16 KiB and nt1 is +1 KiB, so the GEMM loop
//     addresses B entirely with immediate offsets from one base VGPR pair.
//     Also packs obj(3)+box(12)+pad(1) head weights as one 16x256 B matrix.
// ---------------------------------------------------------------------------
__global__ __launch_bounds__(256) void rpn_pack(
    const float* __restrict__ cw, const float* __restrict__ ow,
    const float* __restrict__ bw,
    unsigned short* __restrict__ wpack, unsigned short* __restrict__ hpack) {
  int idx = blockIdx.x * 256 + threadIdx.x;
  if (idx < 589824) {
    int e = idx & 15, l = (idx >> 4) & 31;
    int g = idx >> 9;
    int nt = g & 15, s = g >> 4;
    int oc = nt * 16 + (l & 15);
    int K = s * 32 + ((l >> 4) << 3) + ((e < 8) ? e : (e + 8));
    int ic = K & 255, khw = K >> 8;
    wpack[idx] = f2bf(cw[((size_t)oc * 256 + ic) * 9 + khw]);
  } else if (idx < 589824 + 4096) {
    int j = idx - 589824;
    int e = j & 15, l = (j >> 4) & 31, s = j >> 9;
    int n = l & 15;
    int ic = s * 32 + ((l >> 4) << 3) + ((e < 8) ? e : (e + 8));
    float v = 0.f;
    if (n < 3)       v = ow[n * 256 + ic];
    else if (n < 15) v = bw[(n - 3) * 256 + ic];
    hpack[j] = f2bf(v);
  }
}

__global__ void rpn_zero(unsigned int* __restrict__ hist,
                         unsigned int* __restrict__ meta) {
  int i = blockIdx.x * 256 + threadIdx.x;
  if (i < 4096) hist[i] = 0u;
  if (i < 8)    meta[i] = 0u;
}

// ---------------------------------------------------------------------------
// K1: fused stem 3x3 conv (implicit GEMM, bf16 WMMA) + ReLU + 1x1 heads +
//     anchors + delta decode + sigmoid scores.
//     Block = 256 thr (8 waves), one 16-pixel M-tile, all 16 N-tiles.
// ---------------------------------------------------------------------------
__global__ __launch_bounds__(256) void rpn_conv(
    const float* __restrict__ f3, const float* __restrict__ f4,
    const float* __restrict__ f5,
    const float* __restrict__ conv_b, const float* __restrict__ obj_b,
    const float* __restrict__ box_b,
    const unsigned short* __restrict__ wpack,
    const unsigned short* __restrict__ hpack,
    float* __restrict__ scores, float* __restrict__ out) {
  // A-fragment staging: 36 k-steps * (32 lanes * 32B + bank pad) = 41,472 B
  __shared__ unsigned char sA[36 * 1152];
  // head A-fragments: 8 k-steps
  __shared__ unsigned char sH[8 * 1152];
  __shared__ float sHead[256];

  const int tid = threadIdx.x;
  const int bt = blockIdx.x;
  int lvl, t0;
  if (bt < 950)       { lvl = 0; t0 = bt; }
  else if (bt < 1188) { lvl = 1; t0 = bt - 950; }
  else                { lvl = 2; t0 = bt - 1188; }
  const int H = (lvl == 0) ? 100 : ((lvl == 1) ? 50 : 25);
  const int W = (lvl == 0) ? 152 : ((lvl == 1) ? 76 : 38);
  const int HW = H * W;
  const int pixbase = (lvl == 0) ? 0 : ((lvl == 1) ? 15200 : 19000);
  const float S = (lvl == 0) ? 8.f : ((lvl == 1) ? 16.f : 32.f);
  const float* feat = (lvl == 0) ? f3 : ((lvl == 1) ? f4 : f5);

  const int wave = tid >> 5;
  const int lane = tid & 31;
  const int nt0 = wave * 2, nt1 = nt0 + 1;
  v8f acc0 = {0, 0, 0, 0, 0, 0, 0, 0};
  v8f acc1 = {0, 0, 0, 0, 0, 0, 0, 0};

  // loop-invariant B base: nt0 tile of this wave's lane; all s / nt1 offsets
  // are compile-time immediates from here (s*16384 B, +1024 B for nt1).
  const unsigned short* __restrict__ wbase =
      wpack + ((size_t)nt0 * 32 + lane) * 16;

  // per-thread fixed pixel for im2col gather (A layout: M = lane%16)
  const int m_g = lane & 15;
  const int p_g = t0 * 16 + m_g;
  const bool pv = (p_g < HW);
  const int y_g = pv ? (p_g / W) : 0;
  const int x_g = pv ? (p_g - y_g * W) : 0;
  const int khi = (lane >> 4) << 3;  // upper half-wave holds K+8

  for (int half = 0; half < 2; ++half) {
    __syncthreads();  // previous reads of sA done
    // ---- phase 1: im2col -> LDS A-fragments (bf16), 72 elements/thread ----
    for (int i = 0; i < 72; ++i) {
      const int j = i * 256 + tid;           // j = sl*512 + e*32 + l
      const int e = (j >> 5) & 15;
      const int sl = j >> 9;
      const int s = half * 36 + sl;
      const int kl = khi + ((e < 8) ? e : (e + 8));  // K within 32-chunk
      const int khw = s >> 3;
      const int ic = ((s & 7) << 5) + kl;
      const int iy = y_g + (khw / 3) - 1;
      const int ix = x_g + (khw % 3) - 1;
      float v = 0.f;
      if (pv && iy >= 0 && iy < H && ix >= 0 && ix < W)
        v = feat[(size_t)ic * HW + iy * W + ix];
      *(unsigned short*)(sA + (size_t)sl * 1152 + lane * 32 + (lane >> 3) * 32 + e * 2) =
          f2bf(v);
    }
    __syncthreads();
    // ---- phase 2: 36 k-steps, 2 N-tiles per wave -> 72 WMMAs/half/wave ----
#pragma unroll 6
    for (int sl = 0; sl < 36; ++sl) {
      const int s = half * 36 + sl;
      const v16u* ap =
          (const v16u*)(sA + (size_t)sl * 1152 + lane * 32 + (lane >> 3) * 32);
      v16bf av = __builtin_bit_cast(v16bf, *ap);
      const v16u* bp0 = (const v16u*)(wbase + (size_t)s * 8192);
      const v16u* bp1 = (const v16u*)(wbase + (size_t)s * 8192 + 512);
      v16bf bv0 = __builtin_bit_cast(v16bf, *bp0);
      v16bf bv1 = __builtin_bit_cast(v16bf, *bp1);
      acc0 = __builtin_amdgcn_wmma_f32_16x16x32_bf16(false, av, false, bv0,
                                                     (short)0, acc0, false, false);
      acc1 = __builtin_amdgcn_wmma_f32_16x16x32_bf16(false, av, false, bv1,
                                                     (short)0, acc1, false, false);
    }
  }

  // ---- epilogue: bias+ReLU, restripe stem tile into head A-fragments ----
  {
    const int mo = (lane >= 16) ? 8 : 0;
    const int nl = lane & 15;
    for (int hsel = 0; hsel < 2; ++hsel) {
      const int nt = hsel ? nt1 : nt0;
      const int oc = nt * 16 + nl;
      const float b = conv_b[oc];
      const int sp = oc >> 5;
      const int ocl = oc & 31;
      const int lanehi = (ocl >> 3) & 1;
      const int e = (ocl & 7) + ((ocl & 16) ? 8 : 0);
      for (int r = 0; r < 8; ++r) {
        float v = (hsel ? acc1[r] : acc0[r]) + b;
        v = fmaxf(v, 0.f);
        const int lp = (r + mo) + 16 * lanehi;  // A layout lane = pixel + 16*hi
        *(unsigned short*)(sH + (size_t)sp * 1152 + lp * 32 + (lp >> 3) * 32 + e * 2) =
            f2bf(v);
      }
    }
  }
  __syncthreads();

  // ---- head GEMM on wave 0: [16 pix x 256] x [256 x 16 head-ch] ----
  if (wave == 0) {
    v8f hacc = {0, 0, 0, 0, 0, 0, 0, 0};
    const unsigned short* __restrict__ hbase = hpack + (size_t)lane * 16;
#pragma unroll
    for (int s = 0; s < 8; ++s) {
      const v16u* ap =
          (const v16u*)(sH + (size_t)s * 1152 + lane * 32 + (lane >> 3) * 32);
      const v16u* bp = (const v16u*)(hbase + (size_t)s * 512);
      v16bf av = __builtin_bit_cast(v16bf, *ap);
      v16bf bv = __builtin_bit_cast(v16bf, *bp);
      hacc = __builtin_amdgcn_wmma_f32_16x16x32_bf16(false, av, false, bv,
                                                     (short)0, hacc, false, false);
    }
    const int mo = (lane >= 16) ? 8 : 0;
    const int n = lane & 15;
    for (int r = 0; r < 8; ++r) sHead[(r + mo) * 16 + n] = hacc[r];
  }
  __syncthreads();

  // ---- per-pixel: obj logits, sigmoid scores, anchors, delta decode ----
  if (tid < 16) {
    const int p = t0 * 16 + tid;
    if (p < HW) {
      const int y = p / W;
      const int x = p - y * W;
      const float xc = (y + 0.5f) * S;   // reference: xc from row index
      const float yc = (x + 0.5f) * S;
      const int P = pixbase + p;
      const float area = 16.f * S * S;   // (4*stride)^2
      for (int a = 0; a < 3; ++a) {
        const float logit = sHead[tid * 16 + a] + obj_b[a];
        out[(size_t)P * 3 + a] = logit;
        scores[P * 3 + a] = 1.f / (1.f + expf(-logit));
        const float ar = (a == 0) ? 0.5f : ((a == 1) ? 1.f : 2.f);
        const float w = sqrtf(area / ar);
        const float h = area / w;
        const int AI = P * 3 + a;
        float* anc = out + OUT_ANC + (size_t)AI * 4;
        anc[0] = xc - 0.5f * w; anc[1] = yc - 0.5f * h;
        anc[2] = xc + 0.5f * w; anc[3] = yc + 0.5f * h;
        const float d0 = sHead[tid * 16 + 3 + a * 4 + 0] + box_b[a * 4 + 0];
        const float d1 = sHead[tid * 16 + 3 + a * 4 + 1] + box_b[a * 4 + 1];
        const float d2 = fminf(sHead[tid * 16 + 3 + a * 4 + 2] + box_b[a * 4 + 2],
                               3.33220451017520f);  // log(224/8)
        const float d3 = fminf(sHead[tid * 16 + 3 + a * 4 + 3] + box_b[a * 4 + 3],
                               3.33220451017520f);
        const float bx = xc + w * d0;
        const float by = yc + h * d1;
        const float bh = w * expf(d2);
        const float bw2 = h * expf(d3);
        float* pr = out + OUT_PROP + (size_t)AI * 4;
        if (d0 == 1e-8f) {
          pr[0] = pr[1] = pr[2] = pr[3] = 1e-8f;
        } else {
          pr[0] = bx - 0.5f * bh;  pr[1] = by - 0.5f * bw2;
          pr[2] = bx + 0.5f * bh;  pr[3] = by + 0.5f * bw2;
        }
      }
    }
  }
}

// ---------------------------------------------------------------------------
// Top-k 1000: histogram / threshold / gather / bitonic sort
// ---------------------------------------------------------------------------
__global__ __launch_bounds__(256) void rpn_hist(const float* __restrict__ scores,
                                                unsigned int* __restrict__ hist) {
  int i = blockIdx.x * 256 + threadIdx.x;
  if (i < NANCH) {
    unsigned int k = __float_as_uint(scores[i]);  // sigmoid > 0 -> monotone key
    atomicAdd(&hist[k >> 20], 1u);
  }
}

__global__ void rpn_scan(const unsigned int* __restrict__ hist,
                         unsigned int* __restrict__ meta) {
  if (threadIdx.x != 0 || blockIdx.x != 0) return;
  unsigned int cum = 0, cb = 0, need = TOPK;
  for (int b = 4095; b >= 0; --b) {
    unsigned int c = hist[b];
    if (cum + c >= TOPK) { cb = (unsigned int)b; need = TOPK - cum; break; }
    cum += c;
  }
  meta[0] = cb;
  meta[1] = need;
}

__global__ __launch_bounds__(256) void rpn_gather(
    const float* __restrict__ scores, unsigned int* __restrict__ meta,
    float* __restrict__ pscore, int* __restrict__ pidx) {
  int i = blockIdx.x * 256 + threadIdx.x;
  if (i >= NANCH) return;
  float sc = scores[i];
  unsigned int bin = __float_as_uint(sc) >> 20;
  unsigned int cb = meta[0];
  if (bin > cb) {
    unsigned int slot = atomicAdd(&meta[2], 1u);
    pscore[slot] = sc; pidx[slot] = i;
  } else if (bin == cb) {
    unsigned int t = atomicAdd(&meta[3], 1u);
    if (t < meta[1]) {
      unsigned int slot = atomicAdd(&meta[2], 1u);
      pscore[slot] = sc; pidx[slot] = i;
    }
  }
}

__global__ __launch_bounds__(1024) void rpn_sort(
    const float* __restrict__ pscore, const int* __restrict__ pidx,
    float* __restrict__ out) {
  __shared__ float sk[1024];
  __shared__ int si[1024];
  int t = threadIdx.x;
  if (t < TOPK) { sk[t] = pscore[t]; si[t] = pidx[t]; }
  else          { sk[t] = -1e30f;    si[t] = 0; }
  __syncthreads();
  for (int k = 2; k <= 1024; k <<= 1) {
    for (int j = k >> 1; j > 0; j >>= 1) {
      int ixj = t ^ j;
      if (ixj > t) {
        bool desc = ((t & k) == 0);
        float a = sk[t], b = sk[ixj];
        if (desc ? (a < b) : (a > b)) {
          sk[t] = b; sk[ixj] = a;
          int ti = si[t]; si[t] = si[ixj]; si[ixj] = ti;
        }
      }
      __syncthreads();
    }
  }
  if (t < TOPK) {
    out[OUT_TS + t] = sk[t];
    const float* pr = out + OUT_PROP + (size_t)si[t] * 4;
    float* tb = out + OUT_TB + (size_t)t * 4;
    tb[0] = pr[0]; tb[1] = pr[1]; tb[2] = pr[2]; tb[3] = pr[3];
  }
}

// ---------------------------------------------------------------------------
extern "C" void kernel_launch(void* const* d_in, const int* in_sizes, int n_in,
                              void* d_out, int out_size, void* d_ws,
                              size_t ws_size, hipStream_t stream) {
  (void)in_sizes; (void)n_in; (void)out_size; (void)ws_size;
  const float* f3 = (const float*)d_in[0];
  const float* f4 = (const float*)d_in[1];
  const float* f5 = (const float*)d_in[2];
  const float* cw = (const float*)d_in[3];
  const float* cb = (const float*)d_in[4];
  const float* ow = (const float*)d_in[5];
  const float* ob = (const float*)d_in[6];
  const float* bw = (const float*)d_in[7];
  const float* bb = (const float*)d_in[8];
  float* out = (float*)d_out;
  char* ws = (char*)d_ws;
  unsigned short* wpack = (unsigned short*)(ws + WS_WPACK);
  unsigned short* hpack = (unsigned short*)(ws + WS_HPACK);
  float* scores = (float*)(ws + WS_SCORES);
  unsigned int* hist = (unsigned int*)(ws + WS_HIST);
  unsigned int* meta = (unsigned int*)(ws + WS_META);
  float* pscore = (float*)(ws + WS_PSC);
  int* pidx = (int*)(ws + WS_PIDX);

  rpn_pack<<<2320, 256, 0, stream>>>(cw, ow, bw, wpack, hpack);
  rpn_zero<<<17, 256, 0, stream>>>(hist, meta);
  rpn_conv<<<1248, 256, 0, stream>>>(f3, f4, f5, cb, ob, bb, wpack, hpack,
                                     scores, out);
  rpn_hist<<<(NANCH + 255) / 256, 256, 0, stream>>>(scores, hist);
  rpn_scan<<<1, 1, 0, stream>>>(hist, meta);
  rpn_gather<<<(NANCH + 255) / 256, 256, 0, stream>>>(scores, meta, pscore, pidx);
  rpn_sort<<<1, 1024, 0, stream>>>(pscore, pidx, out);
}
